// ReAttention_52166672777656
// MI455X (gfx1250) — matrix operand order repaired and checked by reading, not compiled, not run on hardware
//
#include <hip/hip_runtime.h>

// ---------------------------------------------------------------------------
// CDNA5 (gfx1250, wave32) implementation of DeepViT-style ReAttention block.
// GEMMs: v_wmma_f32_16x16x32_bf16 (fp32 accumulate).
// Data movement: global_load_async_to_lds_b128 (ASYNCcnt) for pure-copy tiles.
// d_out = [ xo : 8*1024*768 f32 | attn_next : 8*8*1024*1024 f32 ]
// d_ws  = [ q_bf16 : 8*1024*768 u16 | out_bf16 : 8*1024*768 u16 ]  (~24 MB)
// ---------------------------------------------------------------------------

typedef __attribute__((ext_vector_type(16))) __bf16 v16bf;
typedef __attribute__((ext_vector_type(8)))  float  v8f;

__device__ __forceinline__ unsigned short f2bf(float x) {
  unsigned u = __builtin_bit_cast(unsigned, x);
  u += 0x7FFFu + ((u >> 16) & 1u);          // round-to-nearest-even
  return (unsigned short)(u >> 16);
}

// ---- gfx1250 async memory->LDS copy (16B per lane), ASYNCcnt-tracked ------
__device__ __forceinline__ void async_ld_b128(unsigned lds_byte_off,
                                              const void* gaddr) {
  asm volatile("global_load_async_to_lds_b128 %0, %1, off"
               :: "v"(lds_byte_off), "v"(gaddr) : "memory");
}
__device__ __forceinline__ void wait_async0() {
  asm volatile("s_wait_asynccnt 0" ::: "memory");
}
__device__ __forceinline__ unsigned lds_off(const void* p) {
  return (unsigned)(unsigned long long)p;   // LDS aperture: addr[31:0] = offset
}

// A fragment (16xK tile, K=32) from row-major LDS array s[row][ld] (bf16).
// ISA layout: lane m=L%16, grp g=L>>4; half j -> K = (j/8)*16 + g*8 + (j%8).
__device__ __forceinline__ v16bf frag_a16(const unsigned short* s, int row0,
                                          int ld, int k0) {
  int lane = threadIdx.x & 31;
  int m = lane & 15, g = lane >> 4;
  union { v16bf v; unsigned u[8]; } f;
  const unsigned short* base = s + (row0 + m) * ld + k0 + (g << 3);
#pragma unroll
  for (int p = 0; p < 8; ++p) {
    int K = ((p >> 2) << 4) + ((p & 3) << 1);      // dword-pair offset
    f.u[p] = *(const unsigned*)(base + K);
  }
  return f.v;
}

// B fragment (Kx16 tile, K=32), element (k,n) at s[(n0+n)*ld + k0 + k]
// (N-major storage). ISA layout: lane n=L%16, grp g; half j -> K = (j/8)*16 + g*8 + (j%8).
__device__ __forceinline__ v16bf frag_b16(const unsigned short* s, int n0,
                                          int ld, int k0) {
  int lane = threadIdx.x & 31;
  int n = lane & 15, g = lane >> 4;
  union { v16bf v; unsigned u[8]; } f;
  const unsigned short* base = s + (n0 + n) * ld + k0 + (g << 3);
#pragma unroll
  for (int p = 0; p < 8; ++p) f.u[p] = *(const unsigned*)(base + ((p >> 2) << 4) + ((p & 3) << 1));
  return f.v;
}

__device__ __forceinline__ v8f wmma_bf16(v16bf a, v16bf b, v8f c) {
  return __builtin_amdgcn_wmma_f32_16x16x32_bf16(false, a, false, b,
                                                 (short)0, c, false, false);
}

// ---------------------------------------------------------------------------
// Kernel 1: 3x3 SAME conv (3ch -> 3ch), one token (16x16x3) per block.
// Writes q as bf16 (layout [B][N][768], heads = contiguous 96-chunks).
// ---------------------------------------------------------------------------
__global__ void k_conv(const float* __restrict__ x, const float* __restrict__ w,
                       unsigned short* __restrict__ qbf) {
  __shared__ float xs[768];
  __shared__ float wsm[81];
  int token = blockIdx.x;           // b*1024+n, 8192 total
  int tid = threadIdx.x;
  for (int i = tid; i < 768; i += 256) xs[i] = x[(size_t)token * 768 + i];
  if (tid < 81) wsm[tid] = w[tid];
  __syncthreads();
  int i = tid >> 4, j = tid & 15;   // 256 threads = 256 pixels
  float a0 = 0.f, a1 = 0.f, a2 = 0.f;
#pragma unroll
  for (int di = 0; di < 3; ++di) {
    int ii = i + di - 1;
    if (ii < 0 || ii > 15) continue;
#pragma unroll
    for (int dj = 0; dj < 3; ++dj) {
      int jj = j + dj - 1;
      if (jj < 0 || jj > 15) continue;
#pragma unroll
      for (int ci = 0; ci < 3; ++ci) {
        float xv = xs[(ii * 16 + jj) * 3 + ci];
        const float* wp = &wsm[((di * 3 + dj) * 3 + ci) * 3];
        a0 += xv * wp[0]; a1 += xv * wp[1]; a2 += xv * wp[2];
      }
    }
  }
  size_t o = (size_t)token * 768 + (i * 16 + j) * 3;
  qbf[o] = f2bf(a0); qbf[o + 1] = f2bf(a1); qbf[o + 2] = f2bf(a2);
}

// ---------------------------------------------------------------------------
// Kernel 2: S = scale * q q^T per (b,h). Block = 128 rows x 1024 cols.
// q tiles staged with async-to-LDS b128 copies. Raw scaled scores -> attn.
// ---------------------------------------------------------------------------
__global__ void k_scores(const unsigned short* __restrict__ qbf,
                         float* __restrict__ attn) {
  __shared__ unsigned short qa[128 * 96];   // A rows (24 KB)
  __shared__ unsigned short qb[64 * 96];    // B row-chunk (12 KB)
  int rb = blockIdx.x, bh = blockIdx.y;
  int b = bh >> 3, h = bh & 7;
  int tid = threadIdx.x;
  int row0 = rb * 128;
  unsigned qa_base = lds_off(qa), qb_base = lds_off(qb);

  // stage qa: 128 rows x 96 bf16 = 1536 chunks of 16B
  {
    const unsigned short* g0 =
        qbf + ((size_t)(b * 1024 + row0)) * 768 + h * 96;
    for (int c = tid; c < 1536; c += 256) {
      int r = c / 12, c8 = c % 12;
      async_ld_b128(qa_base + c * 16, g0 + (size_t)r * 768 + c8 * 8);
    }
  }
  wait_async0();
  __syncthreads();

  int wave = tid >> 5, lane = tid & 31;
  int wr0 = wave * 16;
  int n = lane & 15, g = lane >> 4;
  v16bf afr0 = frag_a16(qa, wr0, 96, 0);
  v16bf afr1 = frag_a16(qa, wr0, 96, 32);
  v16bf afr2 = frag_a16(qa, wr0, 96, 64);
  const float scale = 0.1020620726159658f;  // 96^-0.5

  for (int ct = 0; ct < 16; ++ct) {
    int m0 = ct * 64;
    __syncthreads();                        // previous qb fully consumed
    {
      const unsigned short* g0 =
          qbf + ((size_t)(b * 1024 + m0)) * 768 + h * 96;
      for (int c = tid; c < 768; c += 256) {  // 64 rows x 12 chunks
        int r = c / 12, c8 = c % 12;
        async_ld_b128(qb_base + c * 16, g0 + (size_t)r * 768 + c8 * 8);
      }
    }
    wait_async0();
    __syncthreads();
#pragma unroll
    for (int t = 0; t < 4; ++t) {
      v8f acc = {0.f, 0.f, 0.f, 0.f, 0.f, 0.f, 0.f, 0.f};
      acc = wmma_bf16(afr0, frag_b16(qb, t * 16, 96, 0), acc);
      acc = wmma_bf16(afr1, frag_b16(qb, t * 16, 96, 32), acc);
      acc = wmma_bf16(afr2, frag_b16(qb, t * 16, 96, 64), acc);
      size_t base =
          ((size_t)bh * 1024 + row0 + wr0 + g * 8) * 1024 + m0 + t * 16 + n;
#pragma unroll
      for (int vr = 0; vr < 8; ++vr)
        attn[base + (size_t)vr * 1024] = acc[vr] * scale;
    }
  }
}

// ---------------------------------------------------------------------------
// Kernel 3 (fused): softmax + re-attention + BN, in place over attn region.
// One block per (b, n): 8 waves, wave h softmaxes row attn[b,h,n,:] into LDS,
// then all threads head-mix + BN and write attn_next rows for all 8 g.
// Saves a full 512 MB round trip vs separate kernels.
// ---------------------------------------------------------------------------
__global__ void k_softmax_reatten(float* __restrict__ attn,
                                  const float* __restrict__ W,
                                  const float* __restrict__ bias,
                                  const float* __restrict__ gamma,
                                  const float* __restrict__ beta,
                                  const float* __restrict__ mean,
                                  const float* __restrict__ var) {
  __shared__ float sm[8 * 1024];            // 32 KB: softmaxed rows, all heads
  int tid = threadIdx.x;
  int wave = tid >> 5, lane = tid & 31;     // wave == head h
  int bn = blockIdx.x;                      // b*1024 + n
  int b = bn >> 10, nn = bn & 1023;

  const size_t rowbase = (((size_t)b * 8 + wave) * 1024 + nn) * 1024;
  float v[32];
#pragma unroll
  for (int i = 0; i < 32; ++i) v[i] = attn[rowbase + lane + i * 32];
  float m = v[0];
#pragma unroll
  for (int i = 1; i < 32; ++i) m = fmaxf(m, v[i]);
#pragma unroll
  for (int off = 16; off > 0; off >>= 1) m = fmaxf(m, __shfl_xor(m, off, 32));
  float s = 0.f;
#pragma unroll
  for (int i = 0; i < 32; ++i) { v[i] = __expf(v[i] - m); s += v[i]; }
#pragma unroll
  for (int off = 16; off > 0; off >>= 1) s += __shfl_xor(s, off, 32);
  float inv = 1.f / s;
#pragma unroll
  for (int i = 0; i < 32; ++i) sm[wave * 1024 + lane + i * 32] = v[i] * inv;
  __syncthreads();

  // head mix + inference BN; block owns rows (b, *, nn, *) exclusively.
  for (int k = 0; k < 4; ++k) {
    int mm = tid + k * 256;
    float a[8];
#pragma unroll
    for (int h = 0; h < 8; ++h) a[h] = sm[h * 1024 + mm];
#pragma unroll
    for (int g = 0; g < 8; ++g) {
      float re = 0.f;
#pragma unroll
      for (int h = 0; h < 8; ++h) re += a[h] * W[h * 8 + g];
      float scg = gamma[g] * rsqrtf(var[g] + 1e-3f);
      float og = (bias[g] - mean[g]) * scg + beta[g];
      attn[(((size_t)b * 8 + g) * 1024 + nn) * 1024 + mm] = re * scg + og;
    }
  }
}

// ---------------------------------------------------------------------------
// Kernel 4: out = attn_next @ q per (b,g). Block = 128 rows x 96 cols.
// A = attn_next (fp32 -> bf16 on the fly), B = q chunk staged transposed so
// both fragments are dword-pair LDS gathers. Output bf16 (for projection).
// ---------------------------------------------------------------------------
__global__ void k_av(const float* __restrict__ attn,
                     const unsigned short* __restrict__ qbf,
                     unsigned short* __restrict__ outbf) {
  __shared__ unsigned short As[128 * 64];   // 16 KB
  __shared__ unsigned short Bt[96 * 64];    // 12 KB, [d][m]
  int rb = blockIdx.x, bh = blockIdx.y;
  int b = bh >> 3, hg = bh & 7;
  int tid = threadIdx.x;
  int row0 = rb * 128;
  int wave = tid >> 5, lane = tid & 31;
  int wr0 = wave * 16;
  int n = lane & 15, g = lane >> 4;
  v8f acc[6];
  v8f zero = {0.f, 0.f, 0.f, 0.f, 0.f, 0.f, 0.f, 0.f};
#pragma unroll
  for (int t = 0; t < 6; ++t) acc[t] = zero;
  for (int mc = 0; mc < 16; ++mc) {
    int m0 = mc * 64;
    __syncthreads();
    for (int i = tid; i < 128 * 64; i += 256) {
      int r = i >> 6, ml = i & 63;
      As[i] = f2bf(attn[((size_t)bh * 1024 + row0 + r) * 1024 + m0 + ml]);
    }
    for (int i = tid; i < 96 * 64; i += 256) {
      int d = i >> 6, ml = i & 63;  // transpose: Bt[d][m] = q[m][d]
      Bt[i] = qbf[((size_t)(b * 1024 + m0 + ml)) * 768 + hg * 96 + d];
    }
    __syncthreads();
#pragma unroll
    for (int kk = 0; kk < 2; ++kk) {
      v16bf af = frag_a16(As, wr0, 64, kk * 32);
#pragma unroll
      for (int t = 0; t < 6; ++t)
        acc[t] = wmma_bf16(af, frag_b16(Bt, t * 16, 64, kk * 32), acc[t]);
    }
  }
#pragma unroll
  for (int t = 0; t < 6; ++t)
#pragma unroll
    for (int vr = 0; vr < 8; ++vr) {
      int row = row0 + wr0 + g * 8 + vr;
      int col = hg * 96 + t * 16 + n;
      outbf[((size_t)(b * 1024 + row)) * 768 + col] = f2bf(acc[t][vr]);
    }
}

// ---------------------------------------------------------------------------
// Kernel 5: xo = out @ proj_w + proj_b. 8192x768x768, 128x128 block tiles.
// A tiles staged with async-to-LDS; W tiles converted+transposed manually.
// ---------------------------------------------------------------------------
__global__ void k_proj(const unsigned short* __restrict__ outbf,
                       const float* __restrict__ Wp,
                       const float* __restrict__ bias, float* __restrict__ xo) {
  __shared__ unsigned short As[128 * 32];   // 8 KB
  __shared__ unsigned short Bt[128 * 32];   // 8 KB, WT[j][p]
  int cb = blockIdx.x, rb = blockIdx.y;
  int tid = threadIdx.x;
  int row0 = rb * 128, col0 = cb * 128;
  int wave = tid >> 5, lane = tid & 31;
  int wr0 = wave * 16;
  int n = lane & 15, g = lane >> 4;
  unsigned as_base = lds_off(As);
  v8f acc[8];
  v8f zero = {0.f, 0.f, 0.f, 0.f, 0.f, 0.f, 0.f, 0.f};
#pragma unroll
  for (int t = 0; t < 8; ++t) acc[t] = zero;
  for (int pc = 0; pc < 24; ++pc) {
    int p0 = pc * 32;
    __syncthreads();
    {  // async-stage A: 128 rows x 32 bf16 = 512 chunks of 16B
      const unsigned short* gA = outbf + (size_t)row0 * 768 + p0;
      for (int c = tid; c < 512; c += 256) {
        int r = c >> 2, c8 = c & 3;
        async_ld_b128(as_base + c * 16, gA + (size_t)r * 768 + c8 * 8);
      }
    }
    for (int i = tid; i < 128 * 32; i += 256) {
      int r = i >> 5, pl = i & 31;
      Bt[i] = f2bf(Wp[((size_t)(p0 + pl)) * 768 + col0 + r]);  // transpose
    }
    wait_async0();
    __syncthreads();
    v16bf af = frag_a16(As, wr0, 32, 0);
#pragma unroll
    for (int t = 0; t < 8; ++t)
      acc[t] = wmma_bf16(af, frag_b16(Bt, t * 16, 32, 0), acc[t]);
  }
#pragma unroll
  for (int t = 0; t < 8; ++t) {
    int col = col0 + t * 16 + n;
    float bb = bias[col];
#pragma unroll
    for (int vr = 0; vr < 8; ++vr) {
      int row = row0 + wr0 + g * 8 + vr;
      xo[(size_t)row * 768 + col] = acc[t][vr] + bb;
    }
  }
}

// ---------------------------------------------------------------------------
extern "C" void kernel_launch(void* const* d_in, const int* in_sizes, int n_in,
                              void* d_out, int out_size, void* d_ws,
                              size_t ws_size, hipStream_t stream) {
  (void)in_sizes; (void)n_in; (void)out_size; (void)ws_size;
  const float* x         = (const float*)d_in[0];
  const float* qconv_w   = (const float*)d_in[1];
  const float* reatten_w = (const float*)d_in[2];
  const float* reatten_b = (const float*)d_in[3];
  const float* bn_gamma  = (const float*)d_in[4];
  const float* bn_beta   = (const float*)d_in[5];
  const float* bn_mean   = (const float*)d_in[6];
  const float* bn_var    = (const float*)d_in[7];
  const float* proj_w    = (const float*)d_in[8];
  const float* proj_b    = (const float*)d_in[9];

  float* xo   = (float*)d_out;
  float* attn = xo + (size_t)8 * 1024 * 768;             // attn_next region

  unsigned short* qbf   = (unsigned short*)d_ws;         // 12 MB
  unsigned short* outbf = qbf + (size_t)8 * 1024 * 768;  // 12 MB

  k_conv           <<<8192,        256, 0, stream>>>(x, qconv_w, qbf);
  k_scores         <<<dim3(8, 64), 256, 0, stream>>>(qbf, attn);
  k_softmax_reatten<<<8192,        256, 0, stream>>>(attn, reatten_w,
                                                     reatten_b, bn_gamma,
                                                     bn_beta, bn_mean, bn_var);
  k_av             <<<dim3(8, 64), 256, 0, stream>>>(attn, qbf, outbf);
  k_proj           <<<dim3(6, 64), 256, 0, stream>>>(outbf, proj_w, proj_b,
                                                     xo);
}
